// MaskedAutoencoderNoise_17197049053494
// MI455X (gfx1250) — matrix operand order repaired and checked by reading, not compiled, not run on hardware
//
#include <hip/hip_runtime.h>

// ---------------------------------------------------------------------------
// Fused per-dim 4-layer MLP:  out[b,d] = W4_d . relu(W3_d . relu(W2_d .
//    relu((mask_d ⊙ W1x_d)ᵀ X[b] + Z[b,d]*W1z_d + b1_d) + b2_d) + b3_d) + b4_d
// B=8192, D=128, H=64.  f16 WMMA (16x16x32) with f32 accumulate.
// ---------------------------------------------------------------------------

typedef __attribute__((ext_vector_type(16))) _Float16 v16h;
typedef __attribute__((ext_vector_type(8)))  _Float16 v8h;
typedef __attribute__((ext_vector_type(8)))  float    v8f;
typedef __attribute__((ext_vector_type(4)))  float    v4f;

#define Bsz 8192
#define Dd  128
#define Hh  64
#define S1  136                 // sW1T row stride in halves (272B = 17*16B, bank-staggered)
#define S2  72                  // sW2T/sW3T/h-stage row stride in halves (144B = 9*16B)
#define GRID_Y 16
#define ROWS_PER_BLOCK (Bsz / GRID_Y)   // 512
#define ROWS_PER_ITER  256              // 8 waves * 32 rows

#define WMMA(a, b, c) \
  __builtin_amdgcn_wmma_f32_16x16x32_f16(false, (a), false, (b), (short)0, (c), false, false)

__device__ __forceinline__ v16h cat16(v8h lo, v8h hi) {
  return __builtin_shufflevector(lo, hi, 0,1,2,3,4,5,6,7,8,9,10,11,12,13,14,15);
}

__device__ __forceinline__ v8f zero8() {
  v8f z;
#pragma unroll
  for (int i = 0; i < 8; ++i) z[i] = 0.0f;
  return z;
}

// A-fragment (16x32 f16, ISA layout) straight from global f32 X.
// lane: m = lane&15 (row), g = lane>>4. halves[0..7]=K kb*32+g*8.., halves[8..15]=K kb*32+16+g*8..
__device__ __forceinline__ v16h load_a_global(const float* __restrict__ X, int row, int kb, int g) {
  const float* p = X + row * Dd + kb * 32 + g * 8;
  v4f f0 = *(const v4f*)(p);
  v4f f1 = *(const v4f*)(p + 4);
  v4f f2 = *(const v4f*)(p + 16);
  v4f f3 = *(const v4f*)(p + 20);
  v16h a;
#pragma unroll
  for (int i = 0; i < 4; ++i) {
    a[i]      = (_Float16)f0[i];
    a[i + 4]  = (_Float16)f1[i];
    a[i + 8]  = (_Float16)f2[i];
    a[i + 12] = (_Float16)f3[i];
  }
  return a;
}

// A-fragment from f16 LDS activation stage (row-major, stride S2 halves).
__device__ __forceinline__ v16h load_a_lds(const _Float16* Hst, int mt, int kb, int m, int g) {
  const _Float16* p = Hst + (mt * 16 + m) * S2 + kb * 32 + g * 8;
  v8h lo = *(const v8h*)(p);        // K kb*32+g*8 .. +7
  v8h hi = *(const v8h*)(p + 16);   // K kb*32+16+g*8 .. +7
  return cat16(lo, hi);
}

// B-fragment (32x16 f16): halves[0..7]=K g*8.., halves[8..15]=K 16+g*8.. of column n.
// Weights stored N-major (Wt[n][k]) so the 8 K-halves of each group are contiguous.
__device__ __forceinline__ v16h load_b_lds(const _Float16* Wt, int stride, int nt, int kb, int n, int g) {
  const _Float16* p = Wt + (nt * 16 + n) * stride + kb * 32 + g * 8;
  v8h lo = *(const v8h*)(p);
  v8h hi = *(const v8h*)(p + 16);
  return cat16(lo, hi);
}

__global__ __launch_bounds__(256)
void fused_mlp_kernel(const float* __restrict__ X,  const float* __restrict__ Z,
                      const float* __restrict__ mask, const float* __restrict__ W1,
                      const float* __restrict__ b1, const float* __restrict__ W2,
                      const float* __restrict__ b2, const float* __restrict__ W3,
                      const float* __restrict__ b3, const float* __restrict__ W4,
                      const float* __restrict__ b4, float* __restrict__ out)
{
  __shared__ _Float16 sW1T[Hh * S1];      // (mask ⊙ W1x)ᵀ : [h][j], f16
  __shared__ _Float16 sW2T[Hh * S2];      // W2ᵀ : [o][h]
  __shared__ _Float16 sW3T[Hh * S2];      // W3ᵀ : [o][h]
  __shared__ float sB1[Hh], sW1z[Hh], sB2[Hh], sB3[Hh], sW4v[Hh];
  __shared__ float sB4;
  __shared__ _Float16 sH[8 * 32 * S2];    // per-wave 32x64 activation stage

  const int d   = blockIdx.x;
  const int tid = threadIdx.x;

  // ---------------- per-d weight prep (f16, transposed, masked) ----------------
#pragma unroll 2
  for (int idx = tid; idx < Dd * Hh; idx += 256) {
    int j = idx >> 6, h = idx & 63;
    float v = mask[j * Dd + d] * W1[(d * 2 * Dd + j) * Hh + h];
    sW1T[h * S1 + j] = (_Float16)v;
  }
#pragma unroll 2
  for (int idx = tid; idx < Hh * Hh; idx += 256) {
    int h = idx >> 6, o = idx & 63;
    sW2T[o * S2 + h] = (_Float16)W2[(d * Hh + h) * Hh + o];
    sW3T[o * S2 + h] = (_Float16)W3[(d * Hh + h) * Hh + o];
  }
  if (tid < Hh) {
    sW1z[tid] = W1[(d * 2 * Dd + Dd + d) * Hh + tid];   // W1[d, 128+d, h]
    sB1[tid]  = b1[d * Hh + tid];
    sB2[tid]  = b2[d * Hh + tid];
    sB3[tid]  = b3[d * Hh + tid];
    sW4v[tid] = W4[d * Hh + tid];
  }
  if (tid == 0) sB4 = b4[d];
  __syncthreads();

  const int wave = tid >> 5;
  const int lane = tid & 31;
  const int m = lane & 15;        // row within 16-row tile (A)
  const int g = lane >> 4;        // lane half
  const int n = m;                // column within 16-col tile (B/C)
  _Float16* Hst = sH + wave * 32 * S2;

  const int blockRowBase = blockIdx.y * ROWS_PER_BLOCK;

#pragma unroll
  for (int it = 0; it < ROWS_PER_BLOCK / ROWS_PER_ITER; ++it) {
    const int rowBase = blockRowBase + it * ROWS_PER_ITER + wave * 32;

    // -------------------- layer 1: (32x128) x (128x64) --------------------
    v8f c1[2][4];
#pragma unroll
    for (int mt = 0; mt < 2; ++mt)
#pragma unroll
      for (int nt = 0; nt < 4; ++nt) c1[mt][nt] = zero8();

#pragma unroll
    for (int kb = 0; kb < 4; ++kb) {
      v16h a0 = load_a_global(X, rowBase + m,      kb, g);
      v16h a1 = load_a_global(X, rowBase + 16 + m, kb, g);
#pragma unroll
      for (int nt = 0; nt < 4; ++nt) {
        v16h bf = load_b_lds(sW1T, S1, nt, kb, n, g);
        c1[0][nt] = WMMA(a0, bf, c1[0][nt]);
        c1[1][nt] = WMMA(a1, bf, c1[1][nt]);
      }
    }

    // epilogue: + Z[b,d]*W1z + b1, ReLU, stage as f16 (C layout: m = v + 8*g)
#pragma unroll
    for (int mt = 0; mt < 2; ++mt) {
      float zv[8];
      const int zr = rowBase + mt * 16 + 8 * g;
#pragma unroll
      for (int v = 0; v < 8; ++v) zv[v] = Z[(zr + v) * Dd + d];
#pragma unroll
      for (int nt = 0; nt < 4; ++nt) {
        float w1z_n = sW1z[nt * 16 + n];
        float b1_n  = sB1[nt * 16 + n];
#pragma unroll
        for (int v = 0; v < 8; ++v) {
          float hv = fmaxf(c1[mt][nt][v] + zv[v] * w1z_n + b1_n, 0.0f);
          Hst[(mt * 16 + v + 8 * g) * S2 + nt * 16 + n] = (_Float16)hv;
        }
      }
    }

    // -------------------- layer 2: (32x64) x (64x64) --------------------
    v8f c2[2][4];
#pragma unroll
    for (int mt = 0; mt < 2; ++mt)
#pragma unroll
      for (int nt = 0; nt < 4; ++nt) c2[mt][nt] = zero8();

#pragma unroll
    for (int kb = 0; kb < 2; ++kb) {
      v16h a0 = load_a_lds(Hst, 0, kb, m, g);
      v16h a1 = load_a_lds(Hst, 1, kb, m, g);
#pragma unroll
      for (int nt = 0; nt < 4; ++nt) {
        v16h bf = load_b_lds(sW2T, S2, nt, kb, n, g);
        c2[0][nt] = WMMA(a0, bf, c2[0][nt]);
        c2[1][nt] = WMMA(a1, bf, c2[1][nt]);
      }
    }
#pragma unroll
    for (int mt = 0; mt < 2; ++mt)
#pragma unroll
      for (int nt = 0; nt < 4; ++nt) {
        float b2_n = sB2[nt * 16 + n];
#pragma unroll
        for (int v = 0; v < 8; ++v) {
          float hv = fmaxf(c2[mt][nt][v] + b2_n, 0.0f);
          Hst[(mt * 16 + v + 8 * g) * S2 + nt * 16 + n] = (_Float16)hv;
        }
      }

    // -------------------- layer 3: (32x64) x (64x64) --------------------
    v8f c3[2][4];
#pragma unroll
    for (int mt = 0; mt < 2; ++mt)
#pragma unroll
      for (int nt = 0; nt < 4; ++nt) c3[mt][nt] = zero8();

#pragma unroll
    for (int kb = 0; kb < 2; ++kb) {
      v16h a0 = load_a_lds(Hst, 0, kb, m, g);
      v16h a1 = load_a_lds(Hst, 1, kb, m, g);
#pragma unroll
      for (int nt = 0; nt < 4; ++nt) {
        v16h bf = load_b_lds(sW3T, S2, nt, kb, n, g);
        c3[0][nt] = WMMA(a0, bf, c3[0][nt]);
        c3[1][nt] = WMMA(a1, bf, c3[1][nt]);
      }
    }

    // ---------- layer 4: fold b3+ReLU, dot with W4, lane-reduce over n ----------
#pragma unroll
    for (int mt = 0; mt < 2; ++mt) {
      float acc[8];
#pragma unroll
      for (int v = 0; v < 8; ++v) acc[v] = 0.0f;
#pragma unroll
      for (int nt = 0; nt < 4; ++nt) {
        float b3_n = sB3[nt * 16 + n];
        float w4_n = sW4v[nt * 16 + n];
#pragma unroll
        for (int v = 0; v < 8; ++v) {
          float hv = fmaxf(c3[mt][nt][v] + b3_n, 0.0f);
          acc[v] += hv * w4_n;
        }
      }
      // butterfly reduce across the 16 n-lanes (masks <=8 keep the two halves apart)
#pragma unroll
      for (int off = 1; off < 16; off <<= 1)
#pragma unroll
        for (int v = 0; v < 8; ++v)
          acc[v] += __shfl_xor(acc[v], off, 32);
      if (n == 0) {
#pragma unroll
        for (int v = 0; v < 8; ++v)
          out[(rowBase + mt * 16 + v + 8 * g) * Dd + d] = acc[v] + sB4;
      }
    }
  }
}

extern "C" void kernel_launch(void* const* d_in, const int* in_sizes, int n_in,
                              void* d_out, int out_size, void* d_ws, size_t ws_size,
                              hipStream_t stream) {
  (void)in_sizes; (void)n_in; (void)d_ws; (void)ws_size; (void)out_size;
  const float* X    = (const float*)d_in[0];
  const float* Z    = (const float*)d_in[1];
  const float* mask = (const float*)d_in[2];
  const float* W1   = (const float*)d_in[3];
  const float* b1   = (const float*)d_in[4];
  const float* W2   = (const float*)d_in[5];
  const float* b2   = (const float*)d_in[6];
  const float* W3   = (const float*)d_in[7];
  const float* b3   = (const float*)d_in[8];
  const float* W4   = (const float*)d_in[9];
  const float* b4   = (const float*)d_in[10];
  float* out = (float*)d_out;

  dim3 grid(Dd, GRID_Y);   // 128 output dims x 16 batch slabs = 2048 workgroups
  fused_mlp_kernel<<<grid, 256, 0, stream>>>(X, Z, mask, W1, b1, W2, b2, W3, b3, W4, b4, out);
}
